// GroupedExperts_18451179504165
// MI455X (gfx1250) — compile-verified, matrix operand order checked by reading, not run on hardware
//
#include <hip/hip_runtime.h>
#include <hip/hip_bf16.h>
#include <stdint.h>

#define N_EXPERTS 16
#define D_MODEL   1024
#define D_FF      2816
#define N_TOKENS  8192
#define TOP_K     2
#define A_TOTAL   (N_TOKENS * TOP_K)          // 16384 flat assignments
#define BM 64
#define BN 64
#define BK 64                                  // two 16x16x32 K-steps per stage
#define KP 72                                  // LDS K-stride: 144B rows, 16B aligned, conflict-free frags
#define PAD_MAX   (A_TOTAL + N_EXPERTS * BM)   // 17408 worst-case padded rows
#define MAX_TILES (PAD_MAX / BM)               // 272 row tiles

typedef __bf16        v16bf __attribute__((ext_vector_type(16)));
typedef float         v8f   __attribute__((ext_vector_type(8)));
typedef unsigned int  u32x4 __attribute__((ext_vector_type(4)));
typedef unsigned int  u32x2 __attribute__((ext_vector_type(2)));
typedef float         f32x4 __attribute__((ext_vector_type(4)));

union FragCast { v16bf v; u32x4 q[2]; };

__device__ __forceinline__ unsigned short f2bf(float f) {
  union { __bf16 h; unsigned short s; } u;
  u.h = (__bf16)f;
  return u.s;
}

// Async DMA: global (bf16, 16B chunk) -> LDS, tracked by ASYNCcnt.
// LDS dest = low 32 bits of generic shared pointer (aperture maps addr[31:0] -> LDS offset).
__device__ __forceinline__ void asyncCopy16(const unsigned short* gsrc, unsigned short* lds) {
  unsigned lo = (unsigned)(uintptr_t)lds;
  unsigned long long ga = (unsigned long long)(uintptr_t)gsrc;
  asm volatile("global_load_async_to_lds_b128 %0, %1, off" :: "v"(lo), "v"(ga) : "memory");
}
__device__ __forceinline__ void asyncWait0() {
  asm volatile("s_wait_asynccnt 0x0" ::: "memory");
}

// ---------------- routing / bookkeeping ----------------

__global__ void kZeroOut(float* out) {
  size_t i = (size_t)blockIdx.x * blockDim.x + threadIdx.x;   // 8192*256 = N_TOKENS*D_MODEL/4
  f32x4 z = {0.f, 0.f, 0.f, 0.f};
  ((f32x4*)out)[i] = z;
}

__global__ void kInitMeta(int* meta) {
  if (threadIdx.x < N_EXPERTS) meta[threadIdx.x] = 0;
}

__global__ void kInitSlots(int* slotTok) {
  int i = blockIdx.x * blockDim.x + threadIdx.x;              // 68*256 = PAD_MAX
  slotTok[i] = -1;
}

__global__ void kCount(const int* __restrict__ eidx, int* meta, int* posOf) {
  int a = blockIdx.x * blockDim.x + threadIdx.x;              // A_TOTAL
  int e = eidx[a];
  posOf[a] = atomicAdd(&meta[e], 1);
}

// meta: [0..15] counts, [16..32] padded offsets, [33] numRowTiles,
//       [64..] tileExpert[272], [352..] tileRow[272]
__global__ void kScan(int* meta) {
  int total = 0, t = 0;
  int* padOff = meta + 16;
  int* tileE  = meta + 64;
  int* tileR  = meta + 352;
  for (int e = 0; e < N_EXPERTS; ++e) {
    padOff[e] = total;
    int pc = (meta[e] + BM - 1) & ~(BM - 1);
    for (int r = 0; r < pc; r += BM) { tileE[t] = e; tileR[t] = total + r; ++t; }
    total += pc;
  }
  padOff[N_EXPERTS] = total;
  meta[33] = t;
}

__global__ void kScatter(const int* __restrict__ eidx, const float* __restrict__ ew,
                         const int* __restrict__ meta, const int* __restrict__ posOf,
                         int* slotTok, float* slotW) {
  int a = blockIdx.x * blockDim.x + threadIdx.x;              // A_TOTAL
  int e = eidx[a];
  int slot = meta[16 + e] + posOf[a];
  slotTok[slot] = a >> 1;                                     // token = a / TOP_K
  slotW[slot]   = ew[a];
}

// gather + fp32 -> bf16 convert of dispatched activations
__global__ void kGather(const float* __restrict__ x, const int* __restrict__ slotTok,
                        unsigned short* xg) {
  int row = blockIdx.x;                                       // PAD_MAX rows
  int c   = threadIdx.x * 4;                                  // 256 thr * 4 = D_MODEL
  int tok = slotTok[row];
  u32x2 packed = {0u, 0u};
  if (tok >= 0) {
    f32x4 v = *(const f32x4*)(x + (size_t)tok * D_MODEL + c);
    packed.x = (unsigned)f2bf(v.x) | ((unsigned)f2bf(v.y) << 16);
    packed.y = (unsigned)f2bf(v.z) | ((unsigned)f2bf(v.w) << 16);
  }
  *(u32x2*)(xg + (size_t)row * D_MODEL + c) = packed;
}

// Convert fp32 weight [e][K][N] -> bf16 transposed [e][N][K] via tiled LDS transpose.
// One-time cost: weights then stream into GEMM LDS tiles as pure 16B async copies.
__global__ void kCvtT(const float* __restrict__ src, unsigned short* __restrict__ dst,
                      int K, int N) {
  __shared__ unsigned short t[32][33];
  int e  = blockIdx.z;
  int n0 = blockIdx.x * 32, k0 = blockIdx.y * 32;
  const float* S = src + (size_t)e * K * N;
  unsigned short* D = dst + (size_t)e * N * K;
  int c  = threadIdx.x & 31;
  int r4 = (threadIdx.x >> 5) * 4;
  #pragma unroll
  for (int i = 0; i < 4; ++i)
    t[r4 + i][c] = f2bf(S[(size_t)(k0 + r4 + i) * N + n0 + c]);
  __syncthreads();
  #pragma unroll
  for (int i = 0; i < 4; ++i)
    D[(size_t)(n0 + r4 + i) * K + k0 + c] = t[c][r4 + i];
}

// ---------------- WMMA fragment load from LDS ----------------
// 16-bit A/B 16x32 wave32 layout: lanes 0-15 -> K{0..7,16..23}; lanes 16-31 -> K{8..15,24..31}
__device__ __forceinline__ v16bf ldsFrag(const unsigned short* base, int row0, int ks, int lane) {
  int r  = row0 + (lane & 15);
  int kb = ks + ((lane >> 4) << 3);
  FragCast f;
  f.q[0] = *(const u32x4*)(base + r * KP + kb);
  f.q[1] = *(const u32x4*)(base + r * KP + kb + 16);
  return f.v;
}

// ---------------- pass 1: h = silu(x@w1) * (x@w2) ----------------

__global__ __launch_bounds__(128)
void kFFN(const unsigned short* __restrict__ xg, const unsigned short* __restrict__ w1T,
          const unsigned short* __restrict__ w2T, const int* __restrict__ meta,
          unsigned short* __restrict__ hbuf) {
  __shared__ unsigned short aT [BM * KP];
  __shared__ unsigned short b1T[BN * KP];
  __shared__ unsigned short b2T[BN * KP];

  int tile = blockIdx.x;
  if (tile >= meta[33]) return;
  int e       = meta[64 + tile];
  int rowBase = meta[352 + tile];
  int n0      = blockIdx.y * BN;

  int tid = threadIdx.x, lane = tid & 31, wave = tid >> 5;
  int wm = wave & 1, wn = wave >> 1;                          // 2x2 waves -> 64x64 tile

  const unsigned short* W1 = w1T + (size_t)e * D_FF * D_MODEL;  // [n][k], k contiguous
  const unsigned short* W2 = w2T + (size_t)e * D_FF * D_MODEL;

  v8f acc1[2][2] = {};
  v8f acc2[2][2] = {};

  for (int k0 = 0; k0 < D_MODEL; k0 += BK) {                  // 16 iterations
    __syncthreads();                                           // LDS reuse safe
    // 64 rows x 64 bf16 per tile = 512 x 16B chunks; 4 per thread per tile, async DMA
    #pragma unroll
    for (int it = 0; it < 4; ++it) {
      int c  = it * 128 + tid;
      int r  = c >> 3;
      int kk = (c & 7) << 3;
      asyncCopy16(xg + (size_t)(rowBase + r) * D_MODEL + k0 + kk, aT  + r * KP + kk);
      asyncCopy16(W1 + (size_t)(n0 + r) * D_MODEL + k0 + kk,     b1T + r * KP + kk);
      asyncCopy16(W2 + (size_t)(n0 + r) * D_MODEL + k0 + kk,     b2T + r * KP + kk);
    }
    asyncWait0();
    __syncthreads();

    #pragma unroll
    for (int ks = 0; ks < BK; ks += 32) {
      v16bf aF[2], b1F[2], b2F[2];
      #pragma unroll
      for (int mi = 0; mi < 2; ++mi) aF[mi] = ldsFrag(aT, wm * 32 + mi * 16, ks, lane);
      #pragma unroll
      for (int ni = 0; ni < 2; ++ni) {
        b1F[ni] = ldsFrag(b1T, wn * 32 + ni * 16, ks, lane);
        b2F[ni] = ldsFrag(b2T, wn * 32 + ni * 16, ks, lane);
      }
      #pragma unroll
      for (int mi = 0; mi < 2; ++mi)
        #pragma unroll
        for (int ni = 0; ni < 2; ++ni) {
          acc1[mi][ni] = __builtin_amdgcn_wmma_f32_16x16x32_bf16(
              false, aF[mi], false, b1F[ni], (short)0, acc1[mi][ni], false, false);
          acc2[mi][ni] = __builtin_amdgcn_wmma_f32_16x16x32_bf16(
              false, aF[mi], false, b2F[ni], (short)0, acc2[mi][ni], false, false);
        }
    }
  }

  // SwiGLU epilogue. C/D layout: lane<16 -> N=lane, M=vgpr j; lane>=16 -> M=8+j, N=lane-16
  int nLane = lane & 15;
  int mOff  = (lane >> 4) << 3;
  #pragma unroll
  for (int mi = 0; mi < 2; ++mi)
    #pragma unroll
    for (int ni = 0; ni < 2; ++ni)
      #pragma unroll
      for (int j = 0; j < 8; ++j) {
        float g  = acc1[mi][ni][j];
        float vv = acc2[mi][ni][j];
        float hv = (g / (1.0f + __expf(-g))) * vv;
        int row = rowBase + wm * 32 + mi * 16 + mOff + j;
        int col = n0 + wn * 32 + ni * 16 + nLane;
        hbuf[(size_t)row * D_FF + col] = f2bf(hv);
      }
}

// ---------------- pass 2: out += weight * (h @ w3) ----------------

__global__ __launch_bounds__(128)
void kOut(const unsigned short* __restrict__ hbuf, const unsigned short* __restrict__ w3T,
          const int* __restrict__ meta, const int* __restrict__ slotTok,
          const float* __restrict__ slotW, float* __restrict__ out) {
  __shared__ unsigned short aT[BM * KP];
  __shared__ unsigned short bT[BN * KP];

  int tile = blockIdx.x;
  if (tile >= meta[33]) return;
  int e       = meta[64 + tile];
  int rowBase = meta[352 + tile];
  int n0      = blockIdx.y * BN;

  int tid = threadIdx.x, lane = tid & 31, wave = tid >> 5;
  int wm = wave & 1, wn = wave >> 1;

  const unsigned short* W3 = w3T + (size_t)e * D_MODEL * D_FF;  // [n=1024][k=2816]

  v8f acc[2][2] = {};

  for (int k0 = 0; k0 < D_FF; k0 += BK) {                     // 44 iterations
    __syncthreads();
    #pragma unroll
    for (int it = 0; it < 4; ++it) {
      int c  = it * 128 + tid;
      int r  = c >> 3;
      int kk = (c & 7) << 3;
      asyncCopy16(hbuf + (size_t)(rowBase + r) * D_FF + k0 + kk, aT + r * KP + kk);
      asyncCopy16(W3 + (size_t)(n0 + r) * D_FF + k0 + kk,        bT + r * KP + kk);
    }
    asyncWait0();
    __syncthreads();

    #pragma unroll
    for (int ks = 0; ks < BK; ks += 32) {
      v16bf aF[2], bF[2];
      #pragma unroll
      for (int mi = 0; mi < 2; ++mi) aF[mi] = ldsFrag(aT, wm * 32 + mi * 16, ks, lane);
      #pragma unroll
      for (int ni = 0; ni < 2; ++ni) bF[ni] = ldsFrag(bT, wn * 32 + ni * 16, ks, lane);
      #pragma unroll
      for (int mi = 0; mi < 2; ++mi)
        #pragma unroll
        for (int ni = 0; ni < 2; ++ni)
          acc[mi][ni] = __builtin_amdgcn_wmma_f32_16x16x32_bf16(
              false, aF[mi], false, bF[ni], (short)0, acc[mi][ni], false, false);
    }
  }

  int nLane = lane & 15;
  int mOff  = (lane >> 4) << 3;
  #pragma unroll
  for (int mi = 0; mi < 2; ++mi)
    #pragma unroll
    for (int ni = 0; ni < 2; ++ni)
      #pragma unroll
      for (int j = 0; j < 8; ++j) {
        int row = rowBase + wm * 32 + mi * 16 + mOff + j;
        int tok = slotTok[row];
        if (tok >= 0) {
          float val = acc[mi][ni][j] * slotW[row];
          int col = n0 + wn * 32 + ni * 16 + nLane;
          atomicAdd(out + (size_t)tok * D_MODEL + col, val);
        }
      }
}

// ---------------- launch ----------------

extern "C" void kernel_launch(void* const* d_in, const int* in_sizes, int n_in,
                              void* d_out, int out_size, void* d_ws, size_t ws_size,
                              hipStream_t stream) {
  const float* x  = (const float*)d_in[0];
  const int*   ei = (const int*)d_in[1];
  const float* ew = (const float*)d_in[2];
  const float* w1 = (const float*)d_in[3];
  const float* w2 = (const float*)d_in[4];
  const float* w3 = (const float*)d_in[5];
  float* out = (float*)d_out;

  char* ws = (char*)d_ws;
  int*            meta    = (int*)(ws + 0);                      // 4 KB meta
  int*            posOf   = (int*)(ws + 4096);                   // int[A_TOTAL]
  int*            slotTok = (int*)(ws + 69632);                  // int[PAD_MAX]
  float*          slotW   = (float*)(ws + 139264);               // float[PAD_MAX]
  unsigned short* xg      = (unsigned short*)(ws + 208896);      // bf16[PAD_MAX][D_MODEL]
  unsigned short* hb      = (unsigned short*)(ws + 35860480ull); // bf16[PAD_MAX][D_FF]
  unsigned short* w1T     = (unsigned short*)(ws + 133902336ull); // bf16[E][D_FF][D_MODEL]
  unsigned short* w2T     = (unsigned short*)(ws + 226177024ull); // bf16[E][D_FF][D_MODEL]
  unsigned short* w3T     = (unsigned short*)(ws + 318451712ull); // bf16[E][D_MODEL][D_FF]

  kZeroOut  <<<8192, 256, 0, stream>>>(out);
  kInitMeta <<<1, 64, 0, stream>>>(meta);
  kInitSlots<<<PAD_MAX / 256, 256, 0, stream>>>(slotTok);
  kCount    <<<A_TOTAL / 256, 256, 0, stream>>>(ei, meta, posOf);
  kScan     <<<1, 1, 0, stream>>>(meta);
  kScatter  <<<A_TOTAL / 256, 256, 0, stream>>>(ei, ew, meta, posOf, slotTok, slotW);
  kGather   <<<PAD_MAX, 256, 0, stream>>>(x, slotTok, xg);

  // one-time fp32 -> bf16 transposed weight conversion
  kCvtT<<<dim3(D_FF / 32, D_MODEL / 32, N_EXPERTS), 256, 0, stream>>>(w1, w1T, D_MODEL, D_FF);
  kCvtT<<<dim3(D_FF / 32, D_MODEL / 32, N_EXPERTS), 256, 0, stream>>>(w2, w2T, D_MODEL, D_FF);
  kCvtT<<<dim3(D_MODEL / 32, D_FF / 32, N_EXPERTS), 256, 0, stream>>>(w3, w3T, D_FF, D_MODEL);

  kFFN<<<dim3(MAX_TILES, D_FF / BN), 128, 0, stream>>>(xg, w1T, w2T, meta, hb);
  kOut<<<dim3(MAX_TILES, D_MODEL / BN), 128, 0, stream>>>(hb, w3T, meta, slotTok, slotW, out);
}
